// JetBlock_22016002359688
// MI455X (gfx1250) — matrix operand order repaired and checked by reading, not compiled, not run on hardware
//
#include <hip/hip_runtime.h>
#include <hip/hip_bf16.h>

// ---------------------------------------------------------------------------
// JetBlock for MI455X (gfx1250): bf16 WMMA GEMMs with TDM (tensor_load_to_lds)
// A-operand staging + fused kern-GEMM/dyn-conv + register-resident gated
// delta-rule scan.
// ---------------------------------------------------------------------------

typedef __attribute__((ext_vector_type(16))) __bf16 v16bf;
typedef __attribute__((ext_vector_type(8)))  __bf16 v8bf;
typedef __attribute__((ext_vector_type(8)))  float  v8f;
typedef __attribute__((ext_vector_type(4)))  unsigned int v4u;
typedef __attribute__((ext_vector_type(8)))  int    v8i32;
typedef __attribute__((ext_vector_type(4)))  int    v4i32;

#define TT     2048            // tokens
#define DD     2048            // model dim
#define NH     16              // heads
#define DKK    128
#define DVV    128
#define WWIN   4               // conv window
#define CONVD  6144            // H*2*DK + H*DV
#define GH     768             // CONVD/8
#define KERNN  (CONVD*WWIN)    // 24576

#define BKP    40              // padded LDS K-stride (bf16 elems) for bank spread

__device__ __forceinline__ float silu_f(float x) {
  return x / (1.f + __expf(-x));
}

// ---------------------------------------------------------------------------
// TDM: DMA a 2D f32 tile (tile1 rows x tile0 elems, row pitch stride0 elems)
// from global memory into LDS at byte offset ldsOff. D# packing per
// CDNA5 ISA. Tracked by TENSORcnt.
// clang-23 toolchain: 6-arg builtin (g0, g1, g2, g3, g4, cpol).
// ---------------------------------------------------------------------------
__device__ __forceinline__ void tdm_load_2d_f32(
    const float* gsrc, unsigned ldsOff, int tile0, int tile1, int stride0)
{
  unsigned long long ga = (unsigned long long)(size_t)gsrc;
  v4u g0;
  g0[0] = 1u;                                            // count=1, user mode
  g0[1] = ldsOff;                                        // lds_addr (bytes)
  g0[2] = (unsigned)ga;                                  // global_addr[31:0]
  g0[3] = (unsigned)((ga >> 32) & 0x01FFFFFFull)         // global_addr[56:32]
          | (2u << 30);                                  // type=2 ("image")
  v8i32 g1;
  g1[0] = (int)(2u << 16);                               // wg_mask=0, data_size=4B
  g1[1] = 0;                                             // abar=0, tdim0.lo16=0
  g1[2] = 0x4000;                                        // tdim0.hi16 (big), tdim1.lo16=0
  g1[3] = (int)(0x4000u | ((unsigned)tile0 << 16));      // tdim1.hi16 (big) | tile_dim0
  g1[4] = tile1;                                         // tile_dim1 | tile_dim2=0
  g1[5] = stride0;                                       // tensor_dim0_stride[31:0]
  g1[6] = 0;                                             // stride0.hi=0, stride1.lo=0
  g1[7] = 0;
  v4i32 z4 = {0, 0, 0, 0};
  v8i32 z8 = {0, 0, 0, 0, 0, 0, 0, 0};
  __builtin_amdgcn_tensor_load_to_lds(g0, g1, z4, z4, z8, 0);
}

// ---------------------------------------------------------------------------
// Shared 128x128 tile mainloop: C_tile += A[rowBase:,:K] * B[:K, colBase:]
// A,B are f32 row-major (lda/ldb).
//   A tile: TDM -> LDS as raw f32 [128][32]; converted to bf16 at frag build.
//   B tile: VALU staging, transposed+converted to bf16 [n][k] in LDS.
// 256 threads = 8 waves arranged 4(M) x 2(N); each wave owns 2x4 WMMA tiles.
// ---------------------------------------------------------------------------
__device__ __forceinline__ void mul_tile128(
    const float* __restrict__ A, const float* __restrict__ B,
    int K, int lda, int ldb, int rowBase, int colBase,
    float* Af, __bf16* Bt, v8f acc[2][4])
{
  const int tid    = threadIdx.x;
  const int lane   = tid & 31;
  const int wid    = tid >> 5;
  const int wm     = wid & 3;          // wave M position (4)
  const int wn     = wid >> 2;         // wave N position (2)
  const int laneHi = (lane >= 16) ? 1 : 0;
  const int l16    = lane & 15;
  const int bk     = tid >> 3;         // B staging: 8 threads/k-row, 16 floats each
  const int bseg   = tid & 7;

  const unsigned afOff = (unsigned)(size_t)(void*)Af;    // LDS byte offset

  for (int kt = 0; kt < K; kt += 32) {
    // ---- A tile via Tensor Data Mover (wave 0 issues; EXEC-independent DMA) ----
    if (wid == 0)
      tdm_load_2d_f32(A + (size_t)rowBase * lda + kt, afOff, 32, 128, lda);

    // ---- stage B tile (32 x 128) transposed -> Bt[n][k], bf16 ----
    {
      const float4* bp =
        (const float4*)(B + (size_t)(kt + bk) * ldb + colBase + bseg * 16);
      #pragma unroll
      for (int i = 0; i < 4; ++i) {
        float4 f = bp[i];
        int n0 = bseg * 16 + i * 4;
        Bt[(n0 + 0) * BKP + bk] = (__bf16)f.x;
        Bt[(n0 + 1) * BKP + bk] = (__bf16)f.y;
        Bt[(n0 + 2) * BKP + bk] = (__bf16)f.z;
        Bt[(n0 + 3) * BKP + bk] = (__bf16)f.w;
      }
    }

    // ---- prefetch next K-tile into L2 (global_prefetch_b8) ----
    if (kt + 32 < K) {
      __builtin_prefetch(B + (size_t)(kt + 32 + bk) * ldb + colBase + bseg * 16, 0, 1);
      __builtin_prefetch(A + (size_t)(rowBase + (tid >> 1)) * lda + kt + 32, 0, 1);
    }

    if (wid == 0)
      __builtin_amdgcn_s_wait_tensorcnt((short)0);       // TDM done before barrier
    __syncthreads();

    // ---- fragments per ISA 16-bit layouts ----
    // A 16x32: lanes 0-15 row M=l, K 0-7 & 16-23; lanes 16-31 K 8-15 & 24-31.
    v16bf afrag[2];
    #pragma unroll
    for (int mt = 0; mt < 2; ++mt) {
      int row = wm * 32 + mt * 16 + l16;
      const float* p = Af + row * 32 + laneHi * 8;
      float4 f0 = *(const float4*)(p);
      float4 f1 = *(const float4*)(p + 4);
      float4 f2 = *(const float4*)(p + 16);
      float4 f3 = *(const float4*)(p + 20);
      v16bf a;
      a[0]  = (__bf16)f0.x; a[1]  = (__bf16)f0.y;
      a[2]  = (__bf16)f0.z; a[3]  = (__bf16)f0.w;
      a[4]  = (__bf16)f1.x; a[5]  = (__bf16)f1.y;
      a[6]  = (__bf16)f1.z; a[7]  = (__bf16)f1.w;
      a[8]  = (__bf16)f2.x; a[9]  = (__bf16)f2.y;
      a[10] = (__bf16)f2.z; a[11] = (__bf16)f2.w;
      a[12] = (__bf16)f3.x; a[13] = (__bf16)f3.y;
      a[14] = (__bf16)f3.z; a[15] = (__bf16)f3.w;
      afrag[mt] = a;
    }
    // B 32x16 (stored transposed): lanes 0-15 col N=l, K 0-7 & 16-23; lanes 16-31 K 8-15 & 24-31.
    v16bf bfrag[4];
    #pragma unroll
    for (int nt = 0; nt < 4; ++nt) {
      int col = wn * 64 + nt * 16 + l16;
      const __bf16* p = Bt + col * BKP + laneHi * 8;
      union { v16bf v; v8bf h[2]; } fr;
      fr.h[0] = *(const v8bf*)(p);
      fr.h[1] = *(const v8bf*)(p + 16);
      bfrag[nt] = fr.v;
    }

    #pragma unroll
    for (int mt = 0; mt < 2; ++mt)
      #pragma unroll
      for (int nt = 0; nt < 4; ++nt)
        acc[mt][nt] = __builtin_amdgcn_wmma_f32_16x16x32_bf16(
            false, afrag[mt], false, bfrag[nt],
            (short)0, acc[mt][nt], false, false);

    __syncthreads();
  }
}

// ---------------------------------------------------------------------------
// Generic GEMM: C = act(A*B).  ACT: 0 = none, 1 = silu.
// ---------------------------------------------------------------------------
template<int ACT>
__global__ __launch_bounds__(256) void gemm_bf16(
    const float* __restrict__ A, const float* __restrict__ B,
    float* __restrict__ C, int M, int N, int K)
{
  __shared__ __align__(16) float  Af[128 * 32];
  __shared__ __align__(16) __bf16 Bt[128 * BKP];
  (void)M;

  const int rowBase = blockIdx.y * 128;
  const int colBase = blockIdx.x * 128;

  v8f acc[2][4];
  v8f zero = {};
  #pragma unroll
  for (int mt = 0; mt < 2; ++mt)
    #pragma unroll
    for (int nt = 0; nt < 4; ++nt) acc[mt][nt] = zero;

  mul_tile128(A, B, K, K, N, rowBase, colBase, Af, Bt, acc);

  const int tid    = threadIdx.x;
  const int lane   = tid & 31;
  const int wid    = tid >> 5;
  const int wm     = wid & 3, wn = wid >> 2;
  const int laneHi = (lane >= 16) ? 1 : 0;
  const int l16    = lane & 15;

  // D layout: VGPR r, lanes 0-15 -> (M=r, N=lane); lanes 16-31 -> (M=r+8, N=lane-16)
  #pragma unroll
  for (int mt = 0; mt < 2; ++mt)
    #pragma unroll
    for (int nt = 0; nt < 4; ++nt)
      #pragma unroll
      for (int r = 0; r < 8; ++r) {
        int m = rowBase + wm * 32 + mt * 16 + laneHi * 8 + r;
        int n = colBase + wn * 64 + nt * 16 + l16;
        float v = acc[mt][nt][r];
        if (ACT == 1) v = silu_f(v);
        C[(size_t)m * N + n] = v;
      }
}

// ---------------------------------------------------------------------------
// Fused: kern = h*gen_w2 + b2 (never materialized), then per-token dynamic
// causal conv over u with the generated 4-tap kernels, then silu -> uc.
// The WMMA D-layout places the 4 taps (w=0..3) of one channel d in 4 adjacent
// lanes, so the tap-sum is a 4-lane shfl_xor reduction.
// ---------------------------------------------------------------------------
__global__ __launch_bounds__(256) void gemm_conv_fused(
    const float* __restrict__ Hh, const float* __restrict__ Wg2,
    const float* __restrict__ b2, const float* __restrict__ U,
    float* __restrict__ Uc)
{
  __shared__ __align__(16) float  Af[128 * 32];
  __shared__ __align__(16) __bf16 Bt[128 * BKP];

  const int rowBase = blockIdx.y * 128;   // tokens
  const int colBase = blockIdx.x * 128;   // (d,w) channels

  v8f acc[2][4];
  v8f zero = {};
  #pragma unroll
  for (int mt = 0; mt < 2; ++mt)
    #pragma unroll
    for (int nt = 0; nt < 4; ++nt) acc[mt][nt] = zero;

  mul_tile128(Hh, Wg2, GH, GH, KERNN, rowBase, colBase, Af, Bt, acc);

  const int tid    = threadIdx.x;
  const int lane   = tid & 31;
  const int wid    = tid >> 5;
  const int wm     = wid & 3, wn = wid >> 2;
  const int laneHi = (lane >= 16) ? 1 : 0;
  const int l16    = lane & 15;

  #pragma unroll
  for (int mt = 0; mt < 2; ++mt)
    #pragma unroll
    for (int nt = 0; nt < 4; ++nt)
      #pragma unroll
      for (int r = 0; r < 8; ++r) {
        int ml = wm * 32 + mt * 16 + laneHi * 8 + r;
        int nl = wn * 64 + nt * 16 + l16;
        int t  = rowBase + ml;
        int cg = colBase + nl;           // flattened d*W + w
        int w  = cg & 3;
        int d  = cg >> 2;
        float kv = acc[mt][nt][r] + b2[cg];
        int tt = t - (WWIN - 1) + w;     // causal window, oldest -> newest
        float uv = (tt >= 0) ? U[(size_t)tt * CONVD + d] : 0.f;
        float p = kv * uv;
        p += __shfl_xor(p, 1, 32);
        p += __shfl_xor(p, 2, 32);
        if ((l16 & 3) == 0)
          Uc[(size_t)t * CONVD + d] = silu_f(p);
      }
}

// ---------------------------------------------------------------------------
// ba = x * W_ba ; beta = sigmoid(ba[:, :16]) ; g = -softplus(ba[:, 16:])
// One wave per token: lane j computes column j of the 32-wide output.
// ---------------------------------------------------------------------------
__global__ __launch_bounds__(256) void ba_kernel(
    const float* __restrict__ x, const float* __restrict__ Wba,
    float* __restrict__ betab, float* __restrict__ gbuf)
{
  const int lane = threadIdx.x & 31;
  const int t    = blockIdx.x * 8 + (threadIdx.x >> 5);
  const float* xr = x + (size_t)t * DD;
  float a = 0.f;
  for (int k = 0; k < DD; ++k) a += xr[k] * Wba[k * (2 * NH) + lane];
  if (lane < NH) {
    betab[t * NH + lane] = 1.f / (1.f + __expf(-a));
  } else {
    float sp = fmaxf(a, 0.f) + log1pf(__expf(-fabsf(a)));   // stable softplus
    gbuf[t * NH + (lane - NH)] = -sp;
  }
}

// ---------------------------------------------------------------------------
// L2-normalize q (and scale by DK^-0.5) and k, in place inside uc.
// One thread per (t, h).
// ---------------------------------------------------------------------------
__global__ __launch_bounds__(256) void norm_qk(float* __restrict__ Uc)
{
  const int idx = blockIdx.x * blockDim.x + threadIdx.x;  // T*NH threads
  const int t = idx >> 4, h = idx & 15;
  float* qp = Uc + (size_t)t * CONVD + h * DKK;
  float* kp = qp + NH * DKK;
  float sq = 0.f, sk = 0.f;
  for (int i = 0; i < DKK; ++i) { sq += qp[i] * qp[i]; sk += kp[i] * kp[i]; }
  float rq = rsqrtf(sq + 1e-6f) * 0.08838834764831845f;   // * DK^-0.5
  float rk = rsqrtf(sk + 1e-6f);
  for (int i = 0; i < DKK; ++i) { qp[i] *= rq; kp[i] *= rk; }
}

// ---------------------------------------------------------------------------
// Gated delta-rule scan. One block per head; thread v owns state column
// S[:, v] (128 floats, register resident). Sequential over T.
// ---------------------------------------------------------------------------
__global__ __launch_bounds__(128) void delta_scan(
    const float* __restrict__ Uc, const float* __restrict__ gbuf,
    const float* __restrict__ betab, float* __restrict__ O)
{
  const int h = blockIdx.x;      // 0..15
  const int v = threadIdx.x;     // 0..127 (state column)
  __shared__ float kb[DKK];
  __shared__ float qb[DKK];

  float s[DKK];
  #pragma unroll
  for (int i = 0; i < DKK; ++i) s[i] = 0.f;

  for (int t = 0; t < TT; ++t) {
    const float* row = Uc + (size_t)t * CONVD;
    qb[v] = row[h * DKK + v];
    kb[v] = row[NH * DKK + h * DKK + v];
    float vt = row[2 * NH * DKK + h * DVV + v];
    float ge = __expf(gbuf[t * NH + h]);
    float bt = betab[t * NH + h];
    __syncthreads();

    float pred = 0.f;
    #pragma unroll
    for (int k = 0; k < DKK; ++k) { s[k] *= ge; pred += kb[k] * s[k]; }
    float uu = bt * (vt - pred);
    float o = 0.f;
    #pragma unroll
    for (int k = 0; k < DKK; ++k) { s[k] += kb[k] * uu; o += qb[k] * s[k]; }

    O[(size_t)t * (NH * DVV) + h * DVV + v] = o;
    __syncthreads();
  }
}

// ---------------------------------------------------------------------------
// Launch
// ---------------------------------------------------------------------------
extern "C" void kernel_launch(void* const* d_in, const int* in_sizes, int n_in,
                              void* d_out, int out_size, void* d_ws, size_t ws_size,
                              hipStream_t stream)
{
  (void)in_sizes; (void)n_in; (void)out_size; (void)ws_size;

  const float* x    = (const float*)d_in[0];   // [T, D]
  const float* Wqkv = (const float*)d_in[1];   // [D, CONVD]
  const float* Wba  = (const float*)d_in[2];   // [D, 2H]
  const float* g1   = (const float*)d_in[3];   // [CONVD, GH]
  const float* g2   = (const float*)d_in[4];   // [GH, CONVD*W]
  const float* b2   = (const float*)d_in[5];   // [CONVD*W]
  const float* Wo   = (const float*)d_in[6];   // [H*DV, D]
  float* out = (float*)d_out;                  // [T, D]

  // workspace carve-out (all f32)
  float* u     = (float*)d_ws;                      // [T, CONVD] pre-conv qkv
  float* hbuf  = u     + (size_t)TT * CONVD;        // [T, GH]
  float* uc    = hbuf  + (size_t)TT * GH;           // [T, CONVD] post-conv qkv
  float* obuf  = uc    + (size_t)TT * CONVD;        // [T, H*DV]
  float* betab = obuf  + (size_t)TT * DD;           // [T, H]
  float* gbuf  = betab + (size_t)TT * NH;           // [T, H]

  dim3 blk(256);

  // 1) qkv = x * W_qkv
  gemm_bf16<0><<<dim3(CONVD / 128, TT / 128), blk, 0, stream>>>(
      x, Wqkv, u, TT, CONVD, DD);
  // 2) beta / g (independent of 1)
  ba_kernel<<<dim3(TT / 8), blk, 0, stream>>>(x, Wba, betab, gbuf);
  // 3) h = silu(qkv * gen_w1)
  gemm_bf16<1><<<dim3(GH / 128, TT / 128), blk, 0, stream>>>(
      u, g1, hbuf, TT, GH, CONVD);
  // 4) kern = h * gen_w2 + b2  fused with dynamic causal conv + silu
  gemm_conv_fused<<<dim3(KERNN / 128, TT / 128), blk, 0, stream>>>(
      hbuf, g2, b2, u, uc);
  // 5) l2norm q,k
  norm_qk<<<dim3((TT * NH) / 256), blk, 0, stream>>>(uc);
  // 6) gated delta-rule scan
  delta_scan<<<dim3(NH), dim3(128), 0, stream>>>(uc, gbuf, betab, obuf);
  // 7) y = o * W_o
  gemm_bf16<0><<<dim3(DD / 128, TT / 128), blk, 0, stream>>>(
      obuf, Wo, out, TT, DD, NH * DVV);
}
